// LSTM_30073361007081
// MI455X (gfx1250) — matrix operand order repaired and checked by reading, not compiled, not run on hardware
//
#include <hip/hip_runtime.h>

// 2-layer LSTM (H=64), B=1024, S=512. Batch-parallel / time-sequential.
// 64 workgroups x 128 threads: each workgroup owns 16 batch rows; its 4 waves
// split the gate/N dimension (wave w owns hidden columns [16w,16w+16) for all
// 4 gates). All weight B-fragments are register-resident (loaded once from
// global, fp32->fp16); only the h1/h2 recurrence state round-trips through a
// tiny LDS buffer to convert D-fragment layout -> A-fragment layout.
// Elementwise gate math uses v_exp_f32 + v_rcp_f32 (no IEEE div sequences).

typedef _Float16 v16h __attribute__((ext_vector_type(16)));
typedef _Float16 v8h  __attribute__((ext_vector_type(8)));
typedef float    v8f  __attribute__((ext_vector_type(8)));

#define HDIM  64
#define WSTR  72           // padded h-row stride (halfs) -> rows 144B, 16B aligned
#define BTILE 16           // batch rows per workgroup
#define NWAVE 4            // waves per block (one per j-tile)
#define NTHR  (NWAVE * 32)

// sigmoid(x) = rcp(1 + exp(-x))           -> v_exp_f32 + v_rcp_f32
__device__ __forceinline__ float sigm(float x) {
  return __builtin_amdgcn_rcpf(1.0f + __expf(-x));
}
// tanh(x) = (e-1)/(e+1) = 1 - 2*rcp(e+1)  -> v_exp_f32 + v_rcp_f32 + fma
__device__ __forceinline__ float tanh_(float x) {
  float e = __expf(2.0f * x);
  return 1.0f - 2.0f * __builtin_amdgcn_rcpf(e + 1.0f);
}

// A fragment (16x32 f16, M x K) from LDS h buffer: lane L -> row m = L%16,
// half h = L/16; element e -> K = kbase + 16h + e.
__device__ __forceinline__ v16h load_a_frag(const _Float16* hbuf, int m, int h, int kbase) {
  const v8h* lo = (const v8h*)(hbuf + m * WSTR + kbase + 16 * h);
  const v8h* hi = (const v8h*)(hbuf + m * WSTR + kbase + 16 * h + 8);
  v8h l = *lo, u = *hi;
  v16h a;
#pragma unroll
  for (int e = 0; e < 8; ++e) { a[e] = l[e]; a[e + 8] = u[e]; }
  return a;
}

// B fragment (32x16 f16, K x N) straight from global fp32 weights:
// lane L -> column n = nrow (incl. L%16), element e -> K = kbase + 16*(L/16) + e,
// i.e. 16 consecutive floats of W row n (B[k][n] = W[n][k]).
__device__ __forceinline__ v16h load_b_frag_g(const float* __restrict__ W, int nrow, int h, int kbase) {
  const float* p = W + nrow * HDIM + kbase + 16 * h;
  v16h b;
#pragma unroll
  for (int e = 0; e < 16; ++e) b[e] = (_Float16)p[e];
  return b;
}

__global__ void __launch_bounds__(NTHR)
lstm_persistent(const float* __restrict__ xg,     // (B,S)
                const float* __restrict__ Wih1,   // (256,1)
                const float* __restrict__ Whh1,   // (256,64)
                const float* __restrict__ bih1,
                const float* __restrict__ bhh1,
                const float* __restrict__ Wih2,   // (256,64)
                const float* __restrict__ Whh2,   // (256,64)
                const float* __restrict__ bih2,
                const float* __restrict__ bhh2,
                const float* __restrict__ Wlin,   // (1,64)
                const float* __restrict__ blin,   // (1,)
                float* __restrict__ out,          // (B,S)
                int B, int S) {
  __shared__ _Float16 sH1[BTILE * WSTR];
  __shared__ _Float16 sH2[BTILE * WSTR];
  __shared__ float    sX[BTILE];
  __shared__ float    sWlin[HDIM];

  const int tid   = threadIdx.x;
  const int wave  = tid >> 5;    // j-tile index jt = wave
  const int lane  = tid & 31;
  const int half_ = lane >> 4;   // 0 or 1
  const int nl    = lane & 15;

  for (int i = tid; i < BTILE * WSTR; i += NTHR) {
    sH1[i] = (_Float16)0.0f;
    sH2[i] = (_Float16)0.0f;
  }
  for (int i = tid; i < HDIM; i += NTHR) sWlin[i] = Wlin[i];

  // ---- register-resident weight fragments for this wave's 4 gate tiles ----
  v16h bWhh1[4][2], bWih2[4][2], bWhh2[4][2];
  float bias1v[4], bias2v[4], wih1v[4];
#pragma unroll
  for (int g = 0; g < 4; ++g) {
    const int n = 64 * g + 16 * wave + nl;   // gate-row in (256,64) weights
#pragma unroll
    for (int q = 0; q < 2; ++q) {
      bWhh1[g][q] = load_b_frag_g(Whh1, n, half_, 32 * q);
      bWih2[g][q] = load_b_frag_g(Wih2, n, half_, 32 * q);
      bWhh2[g][q] = load_b_frag_g(Whh2, n, half_, 32 * q);
    }
    bias1v[g] = bih1[n] + bhh1[n];
    bias2v[g] = bih2[n] + bhh2[n];
    wih1v[g]  = Wih1[n];
  }
  const float bl    = blin[0];
  const int   mbase = blockIdx.x * BTILE;

  // c-state for this wave's j-tile, C/D fragment layout:
  // [r] -> (m = r + 8*half_, j = 16*wave + nl)
  float c1s[8], c2s[8];
#pragma unroll
  for (int r = 0; r < 8; ++r) { c1s[r] = 0.0f; c2s[r] = 0.0f; }

  __syncthreads();

  for (int t = 0; t < S; ++t) {
    // ---- snapshot h1_prev / h2_prev, stage x_t ----
    if (wave == 0 && lane < BTILE) {
      sX[lane] = xg[(size_t)(mbase + lane) * S + t];
      if (t + 1 < S) __builtin_prefetch(&xg[(size_t)(mbase + lane) * S + t + 1], 0, 0);
    }
    v16h a0  = load_a_frag(sH1, nl, half_, 0);
    v16h a1  = load_a_frag(sH1, nl, half_, 32);
    v16h ah0 = load_a_frag(sH2, nl, half_, 0);
    v16h ah1 = load_a_frag(sH2, nl, half_, 32);
    __syncthreads();   // x visible; everyone holds h1_prev/h2_prev snapshots

    float xs[8];
#pragma unroll
    for (int r = 0; r < 8; ++r) xs[r] = sX[r + 8 * half_];

    // ---- layer 1: gates = h1 @ Whh1^T + x*Wih1 + (bih1+bhh1) ----
    {
      v8f acc[4];
#pragma unroll
      for (int g = 0; g < 4; ++g) {
        v8f c;
#pragma unroll
        for (int r = 0; r < 8; ++r) c[r] = bias1v[g] + xs[r] * wih1v[g];
        c = __builtin_amdgcn_wmma_f32_16x16x32_f16(false, a0, false, bWhh1[g][0], (short)0, c, false, false);
        c = __builtin_amdgcn_wmma_f32_16x16x32_f16(false, a1, false, bWhh1[g][1], (short)0, c, false, false);
        acc[g] = c;
      }
#pragma unroll
      for (int r = 0; r < 8; ++r) {
        float iv = sigm(acc[0][r]);
        float fv = sigm(acc[1][r]);
        float gv = tanh_(acc[2][r]);
        float ov = sigm(acc[3][r]);
        float cn = fv * c1s[r] + iv * gv;
        c1s[r] = cn;
        float hn = ov * tanh_(cn);
        sH1[(r + 8 * half_) * WSTR + 16 * wave + nl] = (_Float16)hn;
      }
    }
    __syncthreads();   // h1_new visible

    // ---- layer 2: gates = h1_new @ Wih2^T + h2 @ Whh2^T + (bih2+bhh2) ----
    {
      v16h ai0 = load_a_frag(sH1, nl, half_, 0);
      v16h ai1 = load_a_frag(sH1, nl, half_, 32);
      v8f acc[4];
#pragma unroll
      for (int g = 0; g < 4; ++g) {
        v8f c;
#pragma unroll
        for (int r = 0; r < 8; ++r) c[r] = bias2v[g];
        c = __builtin_amdgcn_wmma_f32_16x16x32_f16(false, ai0, false, bWih2[g][0], (short)0, c, false, false);
        c = __builtin_amdgcn_wmma_f32_16x16x32_f16(false, ai1, false, bWih2[g][1], (short)0, c, false, false);
        c = __builtin_amdgcn_wmma_f32_16x16x32_f16(false, ah0, false, bWhh2[g][0], (short)0, c, false, false);
        c = __builtin_amdgcn_wmma_f32_16x16x32_f16(false, ah1, false, bWhh2[g][1], (short)0, c, false, false);
        acc[g] = c;
      }
#pragma unroll
      for (int r = 0; r < 8; ++r) {
        float iv = sigm(acc[0][r]);
        float fv = sigm(acc[1][r]);
        float gv = tanh_(acc[2][r]);
        float ov = sigm(acc[3][r]);
        float cn = fv * c2s[r] + iv * gv;
        c2s[r] = cn;
        float hn = ov * tanh_(cn);
        sH2[(r + 8 * half_) * WSTR + 16 * wave + nl] = (_Float16)hn;
      }
    }
    __syncthreads();   // h2_new visible

    // ---- out = h2 @ Wlin^T + blin ----
    if (wave == 0 && lane < BTILE) {
      float o = bl;
      const _Float16* hr = sH2 + lane * WSTR;
#pragma unroll
      for (int k = 0; k < HDIM; ++k) o += (float)hr[k] * sWlin[k];
      out[(size_t)(mbase + lane) * S + t] = o;
    }
  }
}

extern "C" void kernel_launch(void* const* d_in, const int* in_sizes, int n_in,
                              void* d_out, int out_size, void* d_ws, size_t ws_size,
                              hipStream_t stream) {
  const float* x    = (const float*)d_in[0];
  const float* Wih1 = (const float*)d_in[1];
  const float* Whh1 = (const float*)d_in[2];
  const float* bih1 = (const float*)d_in[3];
  const float* bhh1 = (const float*)d_in[4];
  const float* Wih2 = (const float*)d_in[5];
  const float* Whh2 = (const float*)d_in[6];
  const float* bih2 = (const float*)d_in[7];
  const float* bhh2 = (const float*)d_in[8];
  const float* Wlin = (const float*)d_in[9];
  const float* blin = (const float*)d_in[10];
  (void)d_ws; (void)ws_size; (void)n_in; (void)out_size;

  const int B = 1024;                 // fixed by reference
  const int S = in_sizes[0] / B;      // 512

  dim3 grid(B / BTILE);               // 64 blocks -> 64 WGPs busy
  dim3 block(NTHR);                   // 128 threads = 4 waves
  hipLaunchKernelGGL(lstm_persistent, grid, block, 0, stream,
                     x, Wih1, Whh1, bih1, bhh1, Wih2, Whh2, bih2, bhh2,
                     Wlin, blin, (float*)d_out, B, S);
}
